// Attention_52424370815683
// MI455X (gfx1250) — matrix operand order repaired and checked by reading, not compiled, not run on hardware
//
#include <hip/hip_runtime.h>
#include <hip/hip_bf16.h>

typedef __bf16 bf16_t;
typedef __bf16 v16bf __attribute__((ext_vector_type(16)));
typedef __bf16 v8bf  __attribute__((ext_vector_type(8)));
typedef __bf16 v4bf  __attribute__((ext_vector_type(4)));
typedef float  v8f   __attribute__((ext_vector_type(8)));
typedef int    v4i_t __attribute__((ext_vector_type(4)));

#define S_LEN  2048
#define DMODEL 1024
#define NHEAD  16
#define HD     64

// ---- CDNA5 async global->LDS path (ASYNCcnt), with safe fallback ----------
// Builtin signature (from clang): (global v4i*, lds v4i*, imm offset, imm cpol)
#if __has_builtin(__builtin_amdgcn_global_load_async_to_lds_b128) && \
    __has_builtin(__builtin_amdgcn_s_wait_asynccnt)
#define ASYNC_LDS 1
typedef __attribute__((address_space(1))) v4i_t g1v4i_t;
typedef __attribute__((address_space(3))) v4i_t l3v4i_t;
#define ASYNC_COPY_B128(gsrc, ldst, off) \
  __builtin_amdgcn_global_load_async_to_lds_b128((g1v4i_t*)(gsrc), \
                                                 (l3v4i_t*)(ldst), (off), 0)
#define WAIT_ASYNC(n) __builtin_amdgcn_s_wait_asynccnt(n)
#else
#define ASYNC_LDS 0
#define WAIT_ASYNC(n)
#endif

static __device__ __forceinline__ v8f wmma_bf16(v16bf a, v16bf b, v8f c) {
  // D = A(16x32 bf16) * B(32x16 bf16) + C(16x16 f32)
  return __builtin_amdgcn_wmma_f32_16x16x32_bf16(false, a, false, b, (short)0, c,
                                                 false, false);
}
static __device__ __forceinline__ v16bf cat8(v8bf lo, v8bf hi) {
  return __builtin_shufflevector(lo, hi, 0,1,2,3,4,5,6,7,8,9,10,11,12,13,14,15);
}
static __device__ __forceinline__ v8f vzero8() {
  v8f z = {0.f,0.f,0.f,0.f,0.f,0.f,0.f,0.f};
  return z;
}

// ---------------------------------------------------------------------------
// f32 -> bf16 conversion, 4 elements/thread
// ---------------------------------------------------------------------------
__global__ void k_cvt_bf16(const float* __restrict__ in, bf16_t* __restrict__ out,
                           int n4) {
  int i = blockIdx.x * blockDim.x + threadIdx.x;
  if (i >= n4) return;
  const float4 f = reinterpret_cast<const float4*>(in)[i];
  v4bf o;
  o[0] = (bf16_t)f.x; o[1] = (bf16_t)f.y; o[2] = (bf16_t)f.z; o[3] = (bf16_t)f.w;
  reinterpret_cast<v4bf*>(out)[i] = o;
}

// ---------------------------------------------------------------------------
// One 32(M)x64(N)x32(K) step for a wave: 2 A-fragments x 4 B-fragments = 8 WMMA
// ---------------------------------------------------------------------------
static __device__ __forceinline__ void gemm_step(const bf16_t* __restrict__ A,
                                                 int m0, int l16, int half, int k0,
                                                 const bf16_t (*buf)[40],
                                                 v8f (&acc)[2][4]) {
  const bf16_t* ar0 = &A[(size_t)(m0 + l16) * DMODEL + k0];
  const bf16_t* ar1 = ar0 + (size_t)16 * DMODEL;
  __builtin_prefetch(ar0 + 32, 0, 1);
  __builtin_prefetch(ar1 + 32, 0, 1);
  v16bf a0 = cat8(*(const v8bf*)(ar0 + half * 16),
                  *(const v8bf*)(ar0 + 8 + half * 16));
  v16bf a1 = cat8(*(const v8bf*)(ar1 + half * 16),
                  *(const v8bf*)(ar1 + 8 + half * 16));
#pragma unroll
  for (int t = 0; t < 4; ++t) {
    v16bf bfrag = *(const v16bf*)&buf[t * 16 + l16][half * 16];
    acc[0][t] = wmma_bf16(a0, bfrag, acc[0][t]);
    acc[1][t] = wmma_bf16(a1, bfrag, acc[1][t]);
  }
}

// ---------------------------------------------------------------------------
// C[M,N] = A[M,1024] * W[N,1024]^T  (bf16 in, f32 accum)
// block = 128 threads (4 waves); block tile 128(M) x 64(N); wave tile 32 x 64.
// Weight tile (64x32) double-buffered in LDS via async global->LDS DMA.
// mode 0: scatter bf16 into Q/K/V [B,H,S,hd] (Q scaled by 0.125)
// mode 1: f32 out = acc + bias[n]
// ---------------------------------------------------------------------------
__launch_bounds__(128)
__global__ void k_gemm_nt(const bf16_t* __restrict__ A,
                          const bf16_t* __restrict__ W,
                          int N, int mode,
                          bf16_t* __restrict__ Qb, bf16_t* __restrict__ Kb,
                          bf16_t* __restrict__ Vb,
                          const float* __restrict__ bias,
                          float* __restrict__ Cout) {
  __shared__ __align__(128) bf16_t sB[2][64][40];  // [buf][n_local][k(32)+pad]
  const int lane = threadIdx.x & 31;
  const int wave = threadIdx.x >> 5;
  const int half = lane >> 4;
  const int l16  = lane & 15;
  const int m0   = blockIdx.y * 128 + wave * 32;
  const int n0   = blockIdx.x * 64;

  v8f acc[2][4];
#pragma unroll
  for (int s = 0; s < 2; ++s)
#pragma unroll
    for (int t = 0; t < 4; ++t) acc[s][t] = vzero8();

  // staging: each of 128 threads owns a 32B chunk of the 4KB weight tile
  const int snn = threadIdx.x >> 1;          // 0..63
  const int skk = (threadIdx.x & 1) * 16;    // 0 or 16
  const bf16_t* wsrc = &W[(size_t)(n0 + snn) * DMODEL + skk];
  bf16_t* sdst = &sB[0][snn][skk];
  const int bufstride = 64 * 40;             // elements between buffers

  // prologue: stage k-step 0 into buffer 0
#if ASYNC_LDS
  ASYNC_COPY_B128(wsrc, sdst, 0);
  ASYNC_COPY_B128(wsrc, sdst, 16);
#else
  *(v16bf*)sdst = *(const v16bf*)wsrc;
#endif

  const int NSTEP = DMODEL / 32;  // 32
  for (int s = 0; s < NSTEP - 1; ++s) {
    {   // stage tile s+1 into the other buffer (in flight during compute)
      const bf16_t* src = wsrc + (s + 1) * 32;
      bf16_t* dst = sdst + ((s + 1) & 1) * bufstride;
#if ASYNC_LDS
      ASYNC_COPY_B128(src, dst, 0);
      ASYNC_COPY_B128(src, dst, 16);
#else
      *(v16bf*)dst = *(const v16bf*)src;
#endif
    }
    WAIT_ASYNC(2);        // <=2 outstanding: tile s has landed (in-order)
    __syncthreads();      // all waves' portions of tile s visible
    gemm_step(A, m0, l16, half, s * 32, sB[s & 1], acc);
    __syncthreads();      // reads of this buffer done before it is re-staged
  }
  WAIT_ASYNC(0);
  __syncthreads();
  gemm_step(A, m0, l16, half, (NSTEP - 1) * 32, sB[(NSTEP - 1) & 1], acc);

  if (mode == 0) {
#pragma unroll
    for (int sub = 0; sub < 2; ++sub) {
#pragma unroll
      for (int t = 0; t < 4; ++t) {
        int n   = n0 + t * 16 + l16;
        int typ = n >> 10;               // 0=q 1=k 2=v (uniform per block)
        int h   = (n >> 6) & 15;         // uniform per block
        int d   = n & 63;
        bf16_t* dst = (typ == 0) ? Qb : (typ == 1) ? Kb : Vb;
        float sc = (typ == 0) ? 0.125f : 1.0f;   // hd^-0.5 folded into Q
#pragma unroll
        for (int r = 0; r < 8; ++r) {
          int m = m0 + sub * 16 + r + 8 * half;
          int b = m >> 11, ss = m & (S_LEN - 1);
          dst[((size_t)((b * NHEAD + h) * S_LEN + ss)) * HD + d] =
              (bf16_t)(acc[sub][t][r] * sc);
        }
      }
    }
  } else {
#pragma unroll
    for (int sub = 0; sub < 2; ++sub) {
#pragma unroll
      for (int t = 0; t < 4; ++t) {
        int n = n0 + t * 16 + l16;
        float bv = bias[n];
#pragma unroll
        for (int r = 0; r < 8; ++r) {
          int m = m0 + sub * 16 + r + 8 * half;
          Cout[(size_t)m * N + n] = acc[sub][t][r] + bv;
        }
      }
    }
  }
}

// ---------------------------------------------------------------------------
// Flash attention per (b,h). block = 128 threads (4 waves) = 64 queries.
// Key loop in blocks of 32; K tile staged via async DMA, V^T tile transposed
// through VGPRs; P re-shaped C-layout -> A-layout through per-wave LDS.
// ---------------------------------------------------------------------------
__launch_bounds__(128)
__global__ void k_attn(const bf16_t* __restrict__ Qb, const bf16_t* __restrict__ Kb,
                       const bf16_t* __restrict__ Vb, bf16_t* __restrict__ O) {
  __shared__ __align__(128) bf16_t sK[32][72];       // [key][hd + pad]
  __shared__ __align__(128) bf16_t sVT[64][40];      // [d][key + pad]
  __shared__ __align__(128) bf16_t sP[4][16][40];    // per-wave P staging
  const int lane = threadIdx.x & 31;
  const int wave = threadIdx.x >> 5;
  const int half = lane >> 4;
  const int l16  = lane & 15;
  const int bh   = blockIdx.z * NHEAD + blockIdx.y;
  const size_t base = (size_t)bh * S_LEN * HD;
  const int q0 = blockIdx.x * 64 + wave * 16;

  // Q A-fragments for hd=64 (two 32-wide k-steps), held for the whole loop
  const bf16_t* qrow = Qb + base + (size_t)(q0 + l16) * HD;
  v16bf qf0 = cat8(*(const v8bf*)(qrow + half * 16),
                   *(const v8bf*)(qrow + 8 + half * 16));
  v16bf qf1 = cat8(*(const v8bf*)(qrow + 32 + half * 16),
                   *(const v8bf*)(qrow + 40 + half * 16));

  float m_i[8], l_i[8];
  v8f acc[4] = {vzero8(), vzero8(), vzero8(), vzero8()};
#pragma unroll
  for (int r = 0; r < 8; ++r) { m_i[r] = -1e30f; l_i[r] = 0.f; }

  for (int kb = 0; kb < S_LEN; kb += 32) {
    __syncthreads();
    {   // stage K tile 32x64 (row-major): 32B per thread
      int key = threadIdx.x >> 2;
      int kk  = (threadIdx.x & 3) * 16;
      const bf16_t* src = &Kb[base + (size_t)(kb + key) * HD + kk];
      bf16_t* dst = &sK[key][kk];
#if ASYNC_LDS
      ASYNC_COPY_B128(src, dst, 0);
      ASYNC_COPY_B128(src, dst, 16);
#else
      *(v16bf*)dst = *(const v16bf*)src;
#endif
    }
    {   // stage V^T tile 64x32 (transpose through VGPRs)
      int key = threadIdx.x >> 2;
      int d0  = (threadIdx.x & 3) * 16;
      v16bf vv = *(const v16bf*)&Vb[base + (size_t)(kb + key) * HD + d0];
#pragma unroll
      for (int j = 0; j < 16; ++j) sVT[d0 + j][key] = vv[j];
    }
    WAIT_ASYNC(0);
    __syncthreads();

    // scores for 32 keys: two 16x16 f32 tiles, k-dim = hd = 64 (2 steps)
    v8f s0 = vzero8(), s1 = vzero8();
    s0 = wmma_bf16(qf0, *(const v16bf*)&sK[l16][half * 16], s0);
    s0 = wmma_bf16(qf1, *(const v16bf*)&sK[l16][32 + half * 16], s0);
    s1 = wmma_bf16(qf0, *(const v16bf*)&sK[16 + l16][half * 16], s1);
    s1 = wmma_bf16(qf1, *(const v16bf*)&sK[16 + l16][32 + half * 16], s1);

    // online softmax; C-layout row = r + 8*half, col = l16 (half-wave wide)
#pragma unroll
    for (int r = 0; r < 8; ++r) {
      float mx = fmaxf(s0[r], s1[r]);
      mx = fmaxf(mx, __shfl_xor(mx, 1, 32));
      mx = fmaxf(mx, __shfl_xor(mx, 2, 32));
      mx = fmaxf(mx, __shfl_xor(mx, 4, 32));
      mx = fmaxf(mx, __shfl_xor(mx, 8, 32));
      float mnew = fmaxf(m_i[r], mx);
      float corr = __expf(m_i[r] - mnew);
      float p0 = __expf(s0[r] - mnew);
      float p1 = __expf(s1[r] - mnew);
      float rs = p0 + p1;
      rs += __shfl_xor(rs, 1, 32);
      rs += __shfl_xor(rs, 2, 32);
      rs += __shfl_xor(rs, 4, 32);
      rs += __shfl_xor(rs, 8, 32);
      l_i[r] = l_i[r] * corr + rs;
      m_i[r] = mnew;
#pragma unroll
      for (int t = 0; t < 4; ++t) acc[t][r] = acc[t][r] * corr;
      sP[wave][r + 8 * half][l16]      = (bf16_t)p0;
      sP[wave][r + 8 * half][16 + l16] = (bf16_t)p1;
    }
    __syncthreads();

    // P (16x32, A-layout) x V (32x64)
    const bf16_t* prow = &sP[wave][l16][0];
    v16bf pf = cat8(*(const v8bf*)(prow + half * 16),
                    *(const v8bf*)(prow + 8 + half * 16));
#pragma unroll
    for (int t = 0; t < 4; ++t) {
      v16bf vf = *(const v16bf*)&sVT[t * 16 + l16][half * 16];
      acc[t] = wmma_bf16(pf, vf, acc[t]);
    }
  }

  // finalize: O[b*S+s][h*64+d] (bf16, feeds proj GEMM)
#pragma unroll
  for (int r = 0; r < 8; ++r) {
    float inv = 1.0f / l_i[r];
    int q = q0 + r + 8 * half;
    size_t m = (size_t)blockIdx.z * S_LEN + q;
#pragma unroll
    for (int t = 0; t < 4; ++t) {
      int d = t * 16 + l16;
      O[m * DMODEL + (size_t)blockIdx.y * HD + d] = (bf16_t)(acc[t][r] * inv);
    }
  }
}

// ---------------------------------------------------------------------------
extern "C" void kernel_launch(void* const* d_in, const int* in_sizes, int n_in,
                              void* d_out, int out_size, void* d_ws, size_t ws_size,
                              hipStream_t stream) {
  const float* x      = (const float*)d_in[0];
  // d_in[1] = xpos (unused by the reference computation)
  const float* qkv_w  = (const float*)d_in[2];
  const float* proj_w = (const float*)d_in[3];
  const float* proj_b = (const float*)d_in[4];
  float* out = (float*)d_out;

  char* ws = (char*)d_ws;
  const size_t MB = 1024 * 1024;
  bf16_t* xb    = (bf16_t*)(ws + 0 * MB);    // 4096x1024        (8 MiB)
  bf16_t* wqkv  = (bf16_t*)(ws + 8 * MB);    // 3072x1024        (6 MiB)
  bf16_t* wproj = (bf16_t*)(ws + 14 * MB);   // 1024x1024        (2 MiB)
  bf16_t* Q     = (bf16_t*)(ws + 16 * MB);   // [B,H,S,hd]       (8 MiB)
  bf16_t* Kt    = (bf16_t*)(ws + 24 * MB);   // [B,H,S,hd]       (8 MiB)
  bf16_t* V     = (bf16_t*)(ws + 32 * MB);   // [B,H,S,hd]       (8 MiB)
  bf16_t* Ob    = (bf16_t*)(ws + 40 * MB);   // [B*S, D]         (8 MiB)

  const int M = 2 * S_LEN;  // 4096 rows

  // f32 -> bf16 conversions
  {
    int n4 = M * DMODEL / 4;
    k_cvt_bf16<<<(n4 + 255) / 256, 256, 0, stream>>>(x, xb, n4);
    n4 = 3 * DMODEL * DMODEL / 4;
    k_cvt_bf16<<<(n4 + 255) / 256, 256, 0, stream>>>(qkv_w, wqkv, n4);
    n4 = DMODEL * DMODEL / 4;
    k_cvt_bf16<<<(n4 + 255) / 256, 256, 0, stream>>>(proj_w, wproj, n4);
  }

  // QKV projection: [4096,1024] x [3072,1024]^T -> Q/K/V
  {
    dim3 grid(3 * DMODEL / 64, M / 128);
    k_gemm_nt<<<grid, 128, 0, stream>>>(xb, wqkv, 3 * DMODEL, 0,
                                        Q, Kt, V, nullptr, nullptr);
  }

  // flash attention
  {
    dim3 grid(S_LEN / 64, NHEAD, 2);
    k_attn<<<grid, 128, 0, stream>>>(Q, Kt, V, Ob);
  }

  // output projection: [4096,1024] x [1024,1024]^T + bias -> f32 out
  {
    dim3 grid(DMODEL / 64, M / 128);
    k_gemm_nt<<<grid, 128, 0, stream>>>(Ob, wproj, DMODEL, 1,
                                        nullptr, nullptr, nullptr, proj_b, out);
  }
}
